// gumbel_NS_shGLM_75531294867878
// MI455X (gfx1250) — matrix operand order repaired and checked by reading, not compiled, not run on hardware
//
#include <hip/hip_runtime.h>
#include <cstdint>

// Problem constants (from reference)
#define T_DATA 100000
#define KDIM   1000      // E_NO == I_NO
#define SUB_NO 20
#define T_SYN  201
#define TBROWS 128       // t-rows per block in stage 2
#define HALO   (T_SYN - 1)

typedef __attribute__((ext_vector_type(16))) __bf16 v16bf;
typedef __attribute__((ext_vector_type(8)))  float  v8f;
typedef __attribute__((ext_vector_type(4)))  float  f4;

// ---------------------------------------------------------------------------
// bf16 packing helpers for WMMA operand layouts (CDNA5 ISA §7.12.2)
// ---------------------------------------------------------------------------
static __device__ __forceinline__ v16bf pack16(f4 a, f4 b, f4 c, f4 d) {
  v16bf v;
  v[0]=(__bf16)a.x;  v[1]=(__bf16)a.y;  v[2]=(__bf16)a.z;  v[3]=(__bf16)a.w;
  v[4]=(__bf16)b.x;  v[5]=(__bf16)b.y;  v[6]=(__bf16)b.z;  v[7]=(__bf16)b.w;
  v[8]=(__bf16)c.x;  v[9]=(__bf16)c.y;  v[10]=(__bf16)c.z; v[11]=(__bf16)c.w;
  v[12]=(__bf16)d.x; v[13]=(__bf16)d.y; v[14]=(__bf16)d.z; v[15]=(__bf16)d.w;
  return v;
}

// A tile (16x32 bf16): this lane supplies K = kb..kb+7 and kb+16..kb+23.
// A is streamed once from HBM -> non-temporal loads (keep L2 for B / ws).
static __device__ __forceinline__ v16bf load_a_nt(const float* __restrict__ rowp, int kb) {
  const f4* p = reinterpret_cast<const f4*>(rowp + kb);
  f4 x0 = __builtin_nontemporal_load(p);
  f4 x1 = __builtin_nontemporal_load(p + 1);
  f4 x2 = __builtin_nontemporal_load(p + 4);
  f4 x3 = __builtin_nontemporal_load(p + 5);
  return pack16(x0, x1, x2, x3);
}
static __device__ __forceinline__ v16bf load_a_tail(const float* __restrict__ rowp, int kb) {
  v16bf v;
#pragma unroll
  for (int j = 0; j < 8; ++j) {
    int k0 = kb + j, k1 = kb + 16 + j;
    v[j]     = (k0 < KDIM) ? (__bf16)rowp[k0] : (__bf16)0.0f;
    v[8 + j] = (k1 < KDIM) ? (__bf16)rowp[k1] : (__bf16)0.0f;
  }
  return v;
}

#define WMMA_BF16(A, B, C) \
  __builtin_amdgcn_wmma_f32_16x16x32_bf16(false, (A), false, (B), (short)0, (C), false, false)

// ---------------------------------------------------------------------------
// Stage 0: prepack C_syn_e / C_syn_i (f32 [20,1000]) into bf16 B operands in
// the exact (tile, chunk, lane) WMMA layout, zero-padded to K=1024, N=32.
// bq[(tile*32 + chunk)*32 + lane] = 16 bf16 = this lane's B operand half.
//   tile: 0 = E cols 0-15, 1 = E cols 16-31, 2 = I cols 0-15, 3 = I cols 16-31
// ---------------------------------------------------------------------------
__global__ __launch_bounds__(256) void prepack_b_kernel(
    const float* __restrict__ Ce, const float* __restrict__ Ci,
    v16bf* __restrict__ bq) {
  const int gid  = blockIdx.x * 256 + threadIdx.x;   // 4096 total
  const int lane = gid & 31;
  const int c    = (gid >> 5) & 31;                  // K chunk
  const int tile = gid >> 10;                        // 0..3
  if (tile >= 4) return;
  const float* C = (tile < 2) ? Ce : Ci;
  const int n  = ((tile & 1) ? 16 : 0) + (lane & 15);
  const int kb = c * 32 + ((lane >> 4) ? 8 : 0);     // B layout: hi lanes at K+8 (and K+24)
  v16bf v;
#pragma unroll
  for (int j = 0; j < 16; ++j) {
    const int k = kb + j + (j >= 8 ? 8 : 0);
    const float f = (n < SUB_NO && k < KDIM) ? C[n * KDIM + k] : 0.0f;
    v[j] = (__bf16)f;
  }
  bq[gid] = v;
}

// ---------------------------------------------------------------------------
// Stage 1: syn_e = S_e @ C_syn_e.T, syn_i = S_i @ C_syn_i.T   via bf16 WMMA
// One wave per 16-row M tile; N=20 padded to two 16-wide tiles.
// B-tile loads are issued FIRST each chunk: since VMEM loads complete in
// order, the A-cvt chain's loadcnt waits cover the B loads for free.
// Unrolling is explicitly disabled: the steady-state 8-load/16-cvt/4-wmma
// loop is register-lean; unrolling provoked catastrophic scratch spills.
// ---------------------------------------------------------------------------
__global__ __launch_bounds__(256) void gemm_syn_kernel(
    const float* __restrict__ S_e, const float* __restrict__ S_i,
    const v16bf* __restrict__ bq,
    float* __restrict__ out_e, float* __restrict__ out_i) {
  const int wid  = threadIdx.x >> 5;
  const int lane = threadIdx.x & 31;
  const int m    = blockIdx.x * 8 + wid;
  if (m >= T_DATA / 16) return;

  const int hi = lane >> 4;   // 0: lanes 0-15, 1: lanes 16-31
  const int ln = lane & 15;
  const int kA = hi ? 8 : 0;  // A layout: hi half starts at K+8

  const float* ae_row = S_e + (size_t)(m * 16 + ln) * KDIM;
  const float* ai_row = S_i + (size_t)(m * 16 + ln) * KDIM;
  const v16bf* bqe0 = bq + (0 * 32) * 32 + lane;
  const v16bf* bqe1 = bq + (1 * 32) * 32 + lane;
  const v16bf* bqi0 = bq + (2 * 32) * 32 + lane;
  const v16bf* bqi1 = bq + (3 * 32) * 32 + lane;

  v8f acc_e0 = {}, acc_e1 = {}, acc_i0 = {}, acc_i1 = {};

#pragma clang loop unroll(disable)
  for (int chunk = 0; chunk < 31; ++chunk) {   // K = 0..991, full 32-wide chunks
    const int kb = chunk * 32;
    // B loads first: L2-hot, their latency hides behind the A load+cvt chain
    v16bf bE0 = bqe0[chunk * 32];
    v16bf bE1 = bqe1[chunk * 32];
    v16bf bI0 = bqi0[chunk * 32];
    v16bf bI1 = bqi1[chunk * 32];
    __builtin_prefetch(ae_row + kb + 64, 0, 3);   // global_prefetch_b8 on A stream
    __builtin_prefetch(ai_row + kb + 64, 0, 3);
    v16bf aE = load_a_nt(ae_row, kb + kA);
    v16bf aI = load_a_nt(ai_row, kb + kA);
    acc_e0 = WMMA_BF16(aE, bE0, acc_e0);
    acc_e1 = WMMA_BF16(aE, bE1, acc_e1);
    acc_i0 = WMMA_BF16(aI, bI0, acc_i0);
    acc_i1 = WMMA_BF16(aI, bI1, acc_i1);
  }
  { // tail chunk: A K = 992..999 valid (zero-padded); B already padded in bq
    const int kb = 31 * 32;
    v16bf bE0 = bqe0[31 * 32];
    v16bf bE1 = bqe1[31 * 32];
    v16bf bI0 = bqi0[31 * 32];
    v16bf bI1 = bqi1[31 * 32];
    v16bf aE = load_a_tail(ae_row, kb + kA);
    v16bf aI = load_a_tail(ai_row, kb + kA);
    acc_e0 = WMMA_BF16(aE, bE0, acc_e0);
    acc_e1 = WMMA_BF16(aE, bE1, acc_e1);
    acc_i0 = WMMA_BF16(aI, bI0, acc_i0);
    acc_i1 = WMMA_BF16(aI, bI1, acc_i1);
  }

  // C/D layout: VGPR r -> row (m*16 + hi*8 + r), column = ln (tile0) / 16+ln (tile1)
  const int rowbase = m * 16 + hi * 8;
#pragma unroll
  for (int r = 0; r < 8; ++r) {
    const size_t o = (size_t)(rowbase + r) * SUB_NO;
    out_e[o + ln] = acc_e0[r];
    out_i[o + ln] = acc_i0[r];
    if (ln < SUB_NO - 16) {           // columns 16..19 only
      out_e[o + 16 + ln] = acc_e1[r];
      out_i[o + 16 + ln] = acc_i1[r];
    }
  }
}

// ---------------------------------------------------------------------------
// Stage 2: fused depthwise causal conv (201 taps) + dendritic-tree hop.
// Halo of syn_e/syn_i staged into LDS with async global->LDS copies (ASYNCcnt).
// ---------------------------------------------------------------------------
static __device__ __forceinline__ void async_copy_b128(unsigned ldsoff, const float* g) {
  // generic LDS addresses truncate to the raw LDS byte offset (ISA §10.2)
  asm volatile("global_load_async_to_lds_b128 %0, %1, off" :: "v"(ldsoff), "v"(g) : "memory");
}

__global__ __launch_bounds__(256) void conv_tree_kernel(
    const float* __restrict__ syn_e, const float* __restrict__ syn_i,
    const float* __restrict__ w_e, const float* __restrict__ b_e,
    const float* __restrict__ w_i, const float* __restrict__ b_i,
    const float* __restrict__ C_den, const float* __restrict__ W_sub,
    float* __restrict__ out) {
  __shared__ __align__(16) float lds_se[(TBROWS + HALO) * SUB_NO];
  __shared__ __align__(16) float lds_si[(TBROWS + HALO) * SUB_NO];
  __shared__ float lds_syn[TBROWS * SUB_NO];
  __shared__ float lds_expw[SUB_NO];
  __shared__ float lds_cden[SUB_NO * SUB_NO];

  const int tid = threadIdx.x;
  const int t0  = blockIdx.x * TBROWS;

  for (int i = tid; i < SUB_NO * SUB_NO; i += 256) lds_cden[i] = C_den[i];
  if (tid < SUB_NO) lds_expw[tid] = expf(W_sub[tid]);

  // stage halo + tile of syn_e / syn_i into LDS asynchronously
  const long base   = ((long)t0 - HALO) * SUB_NO;        // may be negative (causal pad)
  const long nelems = (long)T_DATA * SUB_NO;             // 2,000,000 (divisible by 4)
  const int  total4 = ((TBROWS + HALO) * SUB_NO) / 4;    // 1640 float4s per array
  for (int i4 = tid; i4 < total4; i4 += 256) {
    const long g = base + (long)i4 * 4;
    const unsigned de = (unsigned)(uintptr_t)&lds_se[i4 * 4];
    const unsigned di = (unsigned)(uintptr_t)&lds_si[i4 * 4];
    if (g >= 0 && g + 4 <= nelems) {
      async_copy_b128(de, syn_e + g);
      async_copy_b128(di, syn_i + g);
    } else {
      const f4 z = {0.f, 0.f, 0.f, 0.f};
      *reinterpret_cast<f4*>(&lds_se[i4 * 4]) = z;
      *reinterpret_cast<f4*>(&lds_si[i4 * 4]) = z;
    }
  }
  asm volatile("s_wait_asynccnt 0x0" ::: "memory");
  __syncthreads();

  // depthwise causal conv: out[t,s] = b_e[s]+b_i[s] + sum_tau w[s,tau]*x[t+tau-200]
  // work id w = s*TBROWS + t_local  -> weight row is wave-uniform
#pragma unroll
  for (int j = 0; j < (TBROWS * SUB_NO) / 256; ++j) {
    const int w  = tid + 256 * j;
    const int s  = w >> 7;            // / TBROWS
    const int tl = w & (TBROWS - 1);
    float acc = b_e[s] + b_i[s];
    const float* we = w_e + s * T_SYN;
    const float* wi = w_i + s * T_SYN;
    const float* pe = &lds_se[tl * SUB_NO + s];
    const float* pi = &lds_si[tl * SUB_NO + s];
    for (int tau = 0; tau < T_SYN; ++tau) {
      acc = fmaf(we[tau], pe[tau * SUB_NO], acc);
      acc = fmaf(wi[tau], pi[tau * SUB_NO], acc);
    }
    lds_syn[tl * SUB_NO + s] = acc;
  }
  __syncthreads();

  // one hop of tree message passing + outputs
#pragma unroll
  for (int j = 0; j < (TBROWS * SUB_NO) / 256; ++j) {
    const int w  = tid + 256 * j;
    const int s  = w >> 7;
    const int tl = w & (TBROWS - 1);
    const int t  = t0 + tl;
    float y = lds_syn[tl * SUB_NO + s];
#pragma unroll
    for (int sp = 0; sp < SUB_NO; ++sp)
      y = fmaf(lds_syn[tl * SUB_NO + sp] * lds_expw[sp], lds_cden[s * SUB_NO + sp], y);
    if (t < T_DATA) {
      out[(size_t)T_DATA + (size_t)t * SUB_NO + s] = y;                              // Y_s_pad (copy 1)
      out[(size_t)T_DATA + (size_t)T_DATA * SUB_NO + (size_t)t * SUB_NO + s] = y;    // Y_s_pad (copy 2)
      if (s == 0) out[t] = y * lds_expw[0];                                          // final_voltage
    }
  }
}

// ---------------------------------------------------------------------------
extern "C" void kernel_launch(void* const* d_in, const int* in_sizes, int n_in,
                              void* d_out, int out_size, void* d_ws, size_t ws_size,
                              hipStream_t stream) {
  const float* S_e     = (const float*)d_in[0];
  const float* S_i     = (const float*)d_in[1];
  const float* C_syn_e = (const float*)d_in[2];
  const float* C_syn_i = (const float*)d_in[3];
  const float* C_den   = (const float*)d_in[4];
  const float* w_e     = (const float*)d_in[5];
  const float* b_e     = (const float*)d_in[6];
  const float* w_i     = (const float*)d_in[7];
  const float* b_i     = (const float*)d_in[8];
  const float* W_sub   = (const float*)d_in[9];
  float* out  = (float*)d_out;

  // d_ws layout: [bq: 4096 x 32B = 128 KB][syn_e: 8 MB][syn_i: 8 MB]
  v16bf* bq   = (v16bf*)d_ws;
  float* ws_e = (float*)d_ws + 32768;                // after 128 KB
  float* ws_i = ws_e + (size_t)T_DATA * SUB_NO;

  prepack_b_kernel<<<16, 256, 0, stream>>>(C_syn_e, C_syn_i, bq);

  const int mtiles = T_DATA / 16;                    // 6250
  const int g1 = (mtiles + 7) / 8;                   // 8 waves per block
  gemm_syn_kernel<<<g1, 256, 0, stream>>>(S_e, S_i, bq, ws_e, ws_i);

  const int g2 = (T_DATA + TBROWS - 1) / TBROWS;     // 782
  conv_tree_kernel<<<g2, 256, 0, stream>>>(ws_e, ws_i, w_e, b_e, w_i, b_i,
                                           C_den, W_sub, out);
}
